// BitSelfAttention_85899345920569
// MI455X (gfx1250) — matrix operand order repaired and checked by reading, not compiled, not run on hardware
//
#include <hip/hip_runtime.h>
#include <hip/hip_bf16.h>

// ---------------------------------------------------------------------------
// Types for CDNA5 WMMA (wave32, 16x16x32 bf16 -> f32)
// ---------------------------------------------------------------------------
typedef __attribute__((ext_vector_type(16))) __bf16 v16bf;
typedef __attribute__((ext_vector_type(8)))  float  v8f;

struct Frag256 { uint4 lo, hi; };

static __device__ __forceinline__ v16bf make_frag(uint4 lo, uint4 hi) {
  Frag256 f; f.lo = lo; f.hi = hi;
  return __builtin_bit_cast(v16bf, f);
}

static __device__ __forceinline__ unsigned short f2bf(float f) {
  unsigned int u = __float_as_uint(f);
  u += 0x7FFFu + ((u >> 16) & 1u);          // round-to-nearest-even
  return (unsigned short)(u >> 16);
}

// ---------------------------------------------------------------------------
// CDNA5 async Global->LDS copy (ASYNCcnt-tracked), 16 bytes per lane.
// GV addressing: 64-bit VGPR address, LDS dest offset in a VGPR.
// Low 32 bits of a generic pointer to __shared__ give the LDS byte address.
// ---------------------------------------------------------------------------
static __device__ __forceinline__ void async_copy_b128(void* lds_dst, const void* gsrc) {
  unsigned lds = (unsigned)(unsigned long long)lds_dst;
  unsigned long long ga = (unsigned long long)gsrc;
  asm volatile("global_load_async_to_lds_b128 %0, %1, off"
               :: "v"(lds), "v"(ga)
               : "memory");
}

static __device__ __forceinline__ void wait_async_le4() {
#if __has_builtin(__builtin_amdgcn_s_wait_asynccnt)
  __builtin_amdgcn_s_wait_asynccnt(4);
#else
  asm volatile("s_wait_asynccnt 0x4" ::: "memory");
#endif
}
static __device__ __forceinline__ void wait_async_0() {
#if __has_builtin(__builtin_amdgcn_s_wait_asynccnt)
  __builtin_amdgcn_s_wait_asynccnt(0);
#else
  asm volatile("s_wait_asynccnt 0x0" ::: "memory");
#endif
}

// ---------------------------------------------------------------------------
// f32 -> bf16 conversion (grid-stride)
// ---------------------------------------------------------------------------
__global__ void cvt_f32_to_bf16(const float* __restrict__ in,
                                unsigned short* __restrict__ out, int n) {
  int i = blockIdx.x * blockDim.x + threadIdx.x;
  int stride = gridDim.x * blockDim.x;
  for (; i < n; i += stride) out[i] = f2bf(in[i]);
}

// ---------------------------------------------------------------------------
// Tiled WMMA GEMM: C[M,N] = A[M,K] @ W[N,K]^T   (A,W bf16; C f32 or bf16)
// Block tile 128x128, 256 threads = 8 waves, each wave computes 32x64.
// K panels of 32 double-buffered in LDS via async Global->LDS copies:
//   issue panel i+1 into buf^1, s_wait_asynccnt<=4 (in-order completion =>
//   panel i landed), barrier, 8 WMMAs per wave on panel i.
// ---------------------------------------------------------------------------
__launch_bounds__(256)
__global__ void gemm_bf16_wmma(const unsigned short* __restrict__ A,
                               const unsigned short* __restrict__ W,
                               void* __restrict__ Cout,
                               int M, int N, int K, int store_bf16) {
  __shared__ __attribute__((aligned(16))) unsigned short As[2][128][40];
  __shared__ __attribute__((aligned(16))) unsigned short Ws[2][128][40];

  const int tid  = threadIdx.x;
  const int wave = tid >> 5;
  const int lane = tid & 31;
  const int lm   = lane & 15;
  const int lh   = lane >> 4;
  const int wm   = wave >> 1;   // 0..3 : M sub-block of 32 rows
  const int wn   = wave & 1;    // 0..1 : N sub-block of 64 cols
  const int blockM = blockIdx.y * 128;
  const int blockN = blockIdx.x * 128;

  const int ldrow  = tid >> 1;         // 0..127
  const int ldhalf = (tid & 1) * 16;   // 0 or 16 (bf16 elements)

  const unsigned short* gArow = A + (size_t)(blockM + ldrow) * K + ldhalf;
  const unsigned short* gWrow = W + (size_t)(blockN + ldrow) * K + ldhalf;

  // Issue one 32-wide K panel (4 x b128 per thread) into buffer `buf`.
  auto issue_panel = [&](int buf, int k0) {
    async_copy_b128(&As[buf][ldrow][ldhalf],     gArow + k0);
    async_copy_b128(&As[buf][ldrow][ldhalf + 8], gArow + k0 + 8);
    async_copy_b128(&Ws[buf][ldrow][ldhalf],     gWrow + k0);
    async_copy_b128(&Ws[buf][ldrow][ldhalf + 8], gWrow + k0 + 8);
  };

  v8f acc[2][4];
#pragma unroll
  for (int i = 0; i < 2; ++i)
#pragma unroll
    for (int j = 0; j < 4; ++j)
#pragma unroll
      for (int r = 0; r < 8; ++r) acc[i][j][r] = 0.0f;

  issue_panel(0, 0);                       // prologue: panel 0 -> buf 0

  int cur = 0;
  for (int k0 = 0; k0 < K; k0 += 32, cur ^= 1) {
    if (k0 + 32 < K) {
      issue_panel(cur ^ 1, k0 + 32);       // prefetch next panel
      wait_async_le4();                    // panel `cur` has landed (in-order)
    } else {
      wait_async_0();
    }
    __syncthreads();                       // all waves' async data visible

    // A fragments: lane holds row lm, K chunks at lh*8 and 16+lh*8
    v16bf afr[2];
#pragma unroll
    for (int i = 0; i < 2; ++i) {
      int r = wm * 32 + i * 16 + lm;
      afr[i] = make_frag(*(const uint4*)&As[cur][r][lh * 8],
                         *(const uint4*)&As[cur][r][16 + lh * 8]);
    }
    // B fragments: lane holds col lm, K chunks at lh*8 and 16+lh*8
    v16bf bfr[4];
#pragma unroll
    for (int j = 0; j < 4; ++j) {
      int n = wn * 64 + j * 16 + lm;
      bfr[j] = make_frag(*(const uint4*)&Ws[cur][n][lh * 8],
                         *(const uint4*)&Ws[cur][n][16 + lh * 8]);
    }
#pragma unroll
    for (int i = 0; i < 2; ++i)
#pragma unroll
      for (int j = 0; j < 4; ++j)
        acc[i][j] = __builtin_amdgcn_wmma_f32_16x16x32_bf16(
            false, afr[i], false, bfr[j], (short)0, acc[i][j], false, false);
    __syncthreads();                       // done reading `cur` before its reuse
  }

  // Epilogue: C layout row = r + 8*lh, col = lm
#pragma unroll
  for (int i = 0; i < 2; ++i)
#pragma unroll
    for (int j = 0; j < 4; ++j)
#pragma unroll
      for (int r = 0; r < 8; ++r) {
        int grow = blockM + wm * 32 + i * 16 + r + 8 * lh;
        int gcol = blockN + wn * 64 + j * 16 + lm;
        float v = acc[i][j][r];
        if (store_bf16)
          ((unsigned short*)Cout)[(size_t)grow * N + gcol] = f2bf(v);
        else
          ((float*)Cout)[(size_t)grow * N + gcol] = v;
      }
}

// ---------------------------------------------------------------------------
// Flash attention (causal, GQA 4:1). One block = (b, head, 128 q rows).
// 8 waves; each wave owns 16 q rows. K/V tiles of 64 staged in LDS.
// ---------------------------------------------------------------------------
#define ATT_T   2048
#define ATT_D   2048
#define ATT_DKV 512
#define ATT_HQ  32
#define ATT_GQ  4
#define ATT_DH  64

__launch_bounds__(256)
__global__ void flash_attn_wmma(const unsigned short* __restrict__ Qb,  // [B*T, D]
                                const unsigned short* __restrict__ Kb,  // [B*T, DKV]
                                const unsigned short* __restrict__ Vb,  // [B*T, DKV]
                                unsigned short* __restrict__ Ctx) {     // [B*T, D]
  __shared__ __attribute__((aligned(16))) unsigned short Ks[64][72];    // [kv][d]
  __shared__ __attribute__((aligned(16))) unsigned short Vt[64][72];    // [d][kv]
  __shared__ __attribute__((aligned(16))) unsigned short Ps[8][16][72]; // per-wave P

  const int tid  = threadIdx.x;
  const int wave = tid >> 5;
  const int lane = tid & 31;
  const int lm   = lane & 15;
  const int lh   = lane >> 4;

  const int qtiles = ATT_T / 128;
  int bid = blockIdx.x;
  const int qt = bid % qtiles;  bid /= qtiles;
  const int h  = bid % ATT_HQ;  bid /= ATT_HQ;
  const int b  = bid;
  const int kh = h / ATT_GQ;

  const int qr0 = qt * 128 + wave * 16;        // wave's first q row (in sequence)

  // Q fragments held in registers for the whole kernel (A-matrix 16x64 -> 2 frags)
  const unsigned short* qrow =
      Qb + (size_t)(b * ATT_T + qr0 + lm) * ATT_D + h * ATT_DH;
  v16bf qf[2];
  qf[0] = make_frag(*(const uint4*)(qrow + lh * 8),
                    *(const uint4*)(qrow + 16 + lh * 8));
  qf[1] = make_frag(*(const uint4*)(qrow + 32 + lh * 8),
                    *(const uint4*)(qrow + 48 + lh * 8));

  float mrow[8], lrow[8];
  v8f ctxa[4];
#pragma unroll
  for (int r = 0; r < 8; ++r) { mrow[r] = -3.0e38f; lrow[r] = 0.0f; }
#pragma unroll
  for (int d = 0; d < 4; ++d)
#pragma unroll
    for (int r = 0; r < 8; ++r) ctxa[d][r] = 0.0f;

  const int nkv = qt * 2 + 2;                  // causal: kv tiles 0 .. qt*128+127
  for (int jt = 0; jt < nkv; ++jt) {
    const int kv0 = jt * 64;

    // --- cooperative load of K tile and V tile (transposed) ---
    {
      const int row   = tid >> 2;              // 0..63 (kv index in tile)
      const int chunk = (tid & 3) * 16;        // 0,16,32,48 (d offset)
      const uint4* gK = (const uint4*)(Kb + (size_t)(b * ATT_T + kv0 + row) * ATT_DKV +
                                       kh * ATT_DH + chunk);
      *(uint4*)&Ks[row][chunk]     = gK[0];
      *(uint4*)&Ks[row][chunk + 8] = gK[1];

      const uint4* gV = (const uint4*)(Vb + (size_t)(b * ATT_T + kv0 + row) * ATT_DKV +
                                       kh * ATT_DH + chunk);
      uint4 v0 = gV[0], v1 = gV[1];
      unsigned int uv[8] = {v0.x, v0.y, v0.z, v0.w, v1.x, v1.y, v1.z, v1.w};
#pragma unroll
      for (int e = 0; e < 8; ++e) {
        Vt[chunk + 2 * e][row]     = (unsigned short)(uv[e] & 0xffffu);
        Vt[chunk + 2 * e + 1][row] = (unsigned short)(uv[e] >> 16);
      }
    }
    __syncthreads();

    // --- S = Q @ K^T (16 x 64), WMMA over 2 K-steps x 4 N-tiles ---
    v8f s[4];
#pragma unroll
    for (int j = 0; j < 4; ++j)
#pragma unroll
      for (int r = 0; r < 8; ++r) s[j][r] = 0.0f;

#pragma unroll
    for (int ks = 0; ks < 2; ++ks)
#pragma unroll
      for (int j = 0; j < 4; ++j) {
        int n = j * 16 + lm;
        v16bf bf = make_frag(*(const uint4*)&Ks[n][ks * 32 + lh * 8],
                             *(const uint4*)&Ks[n][ks * 32 + 16 + lh * 8]);
        s[j] = __builtin_amdgcn_wmma_f32_16x16x32_bf16(
            false, qf[ks], false, bf, (short)0, s[j], false, false);
      }

    // --- scale + causal mask ---
#pragma unroll
    for (int j = 0; j < 4; ++j)
#pragma unroll
      for (int r = 0; r < 8; ++r) {
        float v = s[j][r] * 0.125f;            // 1/sqrt(64)
        int col = kv0 + j * 16 + lm;
        int rq  = qr0 + r + 8 * lh;
        s[j][r] = (col > rq) ? -3.0e38f : v;
      }

    // --- online softmax: row max / sum via 16-lane butterflies ---
#pragma unroll
    for (int r = 0; r < 8; ++r) {
      float vmax = fmaxf(fmaxf(s[0][r], s[1][r]), fmaxf(s[2][r], s[3][r]));
      vmax = fmaxf(vmax, __shfl_xor(vmax, 1));
      vmax = fmaxf(vmax, __shfl_xor(vmax, 2));
      vmax = fmaxf(vmax, __shfl_xor(vmax, 4));
      vmax = fmaxf(vmax, __shfl_xor(vmax, 8));
      float mnew = fmaxf(mrow[r], vmax);
      float corr = __expf(mrow[r] - mnew);
      mrow[r] = mnew;
      float psum = 0.0f;
#pragma unroll
      for (int j = 0; j < 4; ++j) {
        float p = __expf(s[j][r] - mnew);
        s[j][r] = p;
        psum += p;
      }
      psum += __shfl_xor(psum, 1);
      psum += __shfl_xor(psum, 2);
      psum += __shfl_xor(psum, 4);
      psum += __shfl_xor(psum, 8);
      lrow[r] = lrow[r] * corr + psum;
#pragma unroll
      for (int d = 0; d < 4; ++d) ctxa[d][r] *= corr;
    }

    // --- P (C-layout) -> LDS -> A-fragments ---
#pragma unroll
    for (int j = 0; j < 4; ++j)
#pragma unroll
      for (int r = 0; r < 8; ++r)
        Ps[wave][r + 8 * lh][j * 16 + lm] = f2bf(s[j][r]);

    const unsigned short* prow = &Ps[wave][lm][0];
    v16bf pf[2];
    pf[0] = make_frag(*(const uint4*)(prow + lh * 8),
                      *(const uint4*)(prow + 16 + lh * 8));
    pf[1] = make_frag(*(const uint4*)(prow + 32 + lh * 8),
                      *(const uint4*)(prow + 48 + lh * 8));

    // --- ctx += P @ V (V transposed in LDS serves as B operand) ---
#pragma unroll
    for (int ks = 0; ks < 2; ++ks)
#pragma unroll
      for (int d = 0; d < 4; ++d) {
        v16bf bf = make_frag(*(const uint4*)&Vt[d * 16 + lm][ks * 32 + lh * 8],
                             *(const uint4*)&Vt[d * 16 + lm][ks * 32 + 16 + lh * 8]);
        ctxa[d] = __builtin_amdgcn_wmma_f32_16x16x32_bf16(
            false, pf[ks], false, bf, (short)0, ctxa[d], false, false);
      }
    __syncthreads();
  }

  // --- epilogue: normalize and store ctx as bf16 ---
  float inv[8];
#pragma unroll
  for (int r = 0; r < 8; ++r) inv[r] = 1.0f / lrow[r];
#pragma unroll
  for (int d = 0; d < 4; ++d)
#pragma unroll
    for (int r = 0; r < 8; ++r) {
      int grow = b * ATT_T + qr0 + r + 8 * lh;
      int gcol = h * ATT_DH + d * 16 + lm;
      Ctx[(size_t)grow * ATT_D + gcol] = f2bf(ctxa[d][r] * inv[r]);
    }
}

// ---------------------------------------------------------------------------
// Launch
// ---------------------------------------------------------------------------
extern "C" void kernel_launch(void* const* d_in, const int* in_sizes, int n_in,
                              void* d_out, int out_size, void* d_ws, size_t ws_size,
                              hipStream_t stream) {
  (void)in_sizes; (void)n_in; (void)out_size; (void)ws_size;

  constexpr int Bz = 2, T = 2048, D = 2048, DKV = 512;
  constexpr int M = Bz * T;                    // 4096 rows

  const float* x  = (const float*)d_in[0];
  const float* wq = (const float*)d_in[1];
  const float* wk = (const float*)d_in[2];
  const float* wv = (const float*)d_in[3];
  const float* wo = (const float*)d_in[4];
  // d_in[5]/d_in[6] = hq/hk scalars on device; dimensions are fixed, hardcoded.

  unsigned short* ws = (unsigned short*)d_ws;
  size_t off = 0;
  auto alloc = [&](size_t nelem) { unsigned short* p = ws + off; off += nelem; return p; };
  unsigned short* xb  = alloc((size_t)M * D);
  unsigned short* wqb = alloc((size_t)D * D);
  unsigned short* wkb = alloc((size_t)DKV * D);
  unsigned short* wvb = alloc((size_t)DKV * D);
  unsigned short* wob = alloc((size_t)D * D);
  unsigned short* Qb  = alloc((size_t)M * D);
  unsigned short* Kb  = alloc((size_t)M * DKV);
  unsigned short* Vb  = alloc((size_t)M * DKV);
  unsigned short* Ctx = alloc((size_t)M * D);

  // 1) bf16 conversions
  cvt_f32_to_bf16<<<2048, 256, 0, stream>>>(x,  xb,  M * D);
  cvt_f32_to_bf16<<<2048, 256, 0, stream>>>(wq, wqb, D * D);
  cvt_f32_to_bf16<<<1024, 256, 0, stream>>>(wk, wkb, DKV * D);
  cvt_f32_to_bf16<<<1024, 256, 0, stream>>>(wv, wvb, DKV * D);
  cvt_f32_to_bf16<<<2048, 256, 0, stream>>>(wo, wob, D * D);

  // 2) projections: Q = x@Wq^T, K = x@Wk^T, V = x@Wv^T (bf16 out)
  gemm_bf16_wmma<<<dim3(D / 128,   M / 128), 256, 0, stream>>>(xb, wqb, Qb, M, D,   D, 1);
  gemm_bf16_wmma<<<dim3(DKV / 128, M / 128), 256, 0, stream>>>(xb, wkb, Kb, M, DKV, D, 1);
  gemm_bf16_wmma<<<dim3(DKV / 128, M / 128), 256, 0, stream>>>(xb, wvb, Vb, M, DKV, D, 1);

  // 3) causal flash attention with GQA
  flash_attn_wmma<<<dim3(Bz * ATT_HQ * (T / 128)), 256, 0, stream>>>(Qb, Kb, Vb, Ctx);

  // 4) output projection -> f32 d_out
  gemm_bf16_wmma<<<dim3(D / 128, M / 128), 256, 0, stream>>>(Ctx, wob, d_out, M, D, D, 0);
}
